// GraphNet_86217173500113
// MI455X (gfx1250) — matrix operand order repaired and checked by reading, not compiled, not run on hardware
//
#include <hip/hip_runtime.h>
#include <hip/hip_bf16.h>

#define BB 8
#define NPTS 4096
#define KNN 30
#define BN (BB * NPTS)

typedef __attribute__((ext_vector_type(2)))  float  v2f;
typedef __attribute__((ext_vector_type(8)))  float  v8f;
typedef __attribute__((ext_vector_type(16))) __bf16 v16bf;
typedef __attribute__((ext_vector_type(8)))  __bf16 v8bf;

union ABf16 { v16bf v; v8bf h[2]; };

__device__ __forceinline__ v8f wmma_f32x4(v2f a, v2f b, v8f c) {
  return __builtin_amdgcn_wmma_f32_16x16x4_f32(false, a, false, b, (short)0, c, false, false);
}
__device__ __forceinline__ v8f wmma_bf16(v16bf a, v16bf b, v8f c) {
  return __builtin_amdgcn_wmma_f32_16x16x32_bf16(false, a, false, b, (short)0, c, false, false);
}

// LDS byte address (low 32 bits of the flat address; HW adds LDS_BASE for async ops)
__device__ __forceinline__ unsigned lds_addr(const void* p) {
  return (unsigned)(unsigned long long)p;
}
// 16B global -> LDS DMA (tracked by ASYNCcnt, no VGPR round-trip)
__device__ __forceinline__ void async_copy_b128(unsigned lds_byte, const void* gaddr) {
  asm volatile("global_load_async_to_lds_b128 %0, %1, off" ::"v"(lds_byte), "v"(gaddr)
               : "memory");
}
__device__ __forceinline__ void async_wait0() {
  asm volatile("s_wait_asynccnt 0x0" ::: "memory");
}

// ---------------- prep kernels ----------------

__global__ void pad8_sq_kernel(const float* __restrict__ pts, float* __restrict__ xpad,
                               float* __restrict__ sq) {
  int i = blockIdx.x * 256 + threadIdx.x;
  if (i >= BN) return;
  float s = 0.f;
#pragma unroll
  for (int c = 0; c < 6; c++) { float v = pts[i * 6 + c]; s += v * v; xpad[i * 8 + c] = v; }
  xpad[i * 8 + 6] = 0.f;
  xpad[i * 8 + 7] = 0.f;
  sq[i] = s;
}

__global__ void sqnorm64_kernel(const float* __restrict__ x, float* __restrict__ sq) {
  int i = blockIdx.x * 256 + threadIdx.x;
  if (i >= BN) return;
  float s = 0.f;
#pragma unroll
  for (int c = 0; c < 64; c++) { float v = x[i * 64 + c]; s += v * v; }
  sq[i] = s;
}

// dst[n*KP + k] = (k < K) ? src[k*N + n] : 0   (bf16, N-major "transposed" weights)
__global__ void convert_transpose_kernel(const float* __restrict__ src, __bf16* __restrict__ dst,
                                         int Kdim, int KP, int Ndim) {
  int t = blockIdx.x * 256 + threadIdx.x;
  if (t >= Ndim * KP) return;
  int n = t / KP, k = t % KP;
  dst[t] = (k < Kdim) ? (__bf16)src[k * Ndim + n] : (__bf16)0.0f;
}

// ---------------- KNN: f32-WMMA distances + in-register top-30 ----------------
// Block = 128 threads (4 waves): 64 query rows vs all 4096 points.
// Query rows + row sqnorms live in VGPRs; column tiles double-buffered in LDS via
// GLOBAL_LOAD_ASYNC_TO_LDS_B128 so the next tile's DMA overlaps WMMA + selection.
template <int CIN>
__global__ void __launch_bounds__(128) knn_kernel(const float* __restrict__ x,
                                                  const float* __restrict__ sq,
                                                  int* __restrict__ idxout) {
  __shared__ alignas(16) float ldsColA[64 * CIN];
  __shared__ alignas(16) float ldsColB[64 * CIN];
  __shared__ alignas(16) float ldsDist[64 * 65];  // stride 65: no bank conflicts

  const int tid = threadIdx.x;
  const int wave = tid >> 5, lane = tid & 31;
  const int hi = lane >> 4, l15 = lane & 15;
  const int b = blockIdx.y;
  const int rb = blockIdx.x * 64;
  const float* xb = x + (size_t)b * NPTS * CIN;
  const float* sqb = sq + (size_t)b * NPTS;

  // A operands for this wave's 16 rows, kept in registers for the whole kernel
  v2f areg[CIN / 4];
  const int m = wave * 16 + l15;
#pragma unroll
  for (int kk = 0; kk < CIN; kk += 4)
    areg[kk / 4] = *(const v2f*)&xb[(size_t)(rb + m) * CIN + kk + 2 * hi];
  float sqrow[8];
#pragma unroll
  for (int r = 0; r < 8; r++) sqrow[r] = sqb[rb + wave * 16 + r + 8 * hi];

  float bestd[KNN];
  int besti[KNN];
#pragma unroll
  for (int k = 0; k < KNN; k++) { bestd[k] = 3.0e38f; besti[k] = 0; }

  auto issue_tile = [&](int ct, float* dst) {
    const unsigned base = lds_addr(dst);
    const float* g = xb + (size_t)ct * CIN;
    for (int t = tid * 4; t < 64 * CIN; t += 128 * 4)
      async_copy_b128(base + t * 4, g + t);
  };

  issue_tile(0, ldsColA);
  float* cur = ldsColA;
  float* nxt = ldsColB;

  for (int ct = 0; ct < NPTS; ct += 64) {
    async_wait0();
    __syncthreads();                      // tile `ct` resident for all waves
    if (ct + 64 < NPTS) issue_tile(ct + 64, nxt);  // DMA next tile during compute+select

    for (int nt = 0; nt < 4; nt++) {
      v8f acc = {};
      const int n2 = nt * 16 + l15;
#pragma unroll
      for (int kk = 0; kk < CIN; kk += 4) {
        v2f bb = *(const v2f*)&cur[n2 * CIN + kk + 2 * hi];
        acc = wmma_f32x4(areg[kk / 4], bb, acc);
      }
      const float sqc = sqb[ct + n2];
#pragma unroll
      for (int r = 0; r < 8; r++) {
        int mr = wave * 16 + r + 8 * hi;
        ldsDist[mr * 65 + n2] = sqrow[r] + sqc - 2.0f * acc[r];
      }
    }
    __syncthreads();

    // threads 0..63: one row each; sorted-descending list stays in VGPRs
    if (tid < 64) {
      for (int c = 0; c < 64; c++) {
        float d = ldsDist[tid * 65 + c];
        if (d < bestd[0]) {
          bestd[0] = d;
          besti[0] = ct + c;
#pragma unroll
          for (int t = 0; t < KNN - 1; t++) {
            if (bestd[t] < bestd[t + 1]) {
              float td = bestd[t]; bestd[t] = bestd[t + 1]; bestd[t + 1] = td;
              int ti = besti[t]; besti[t] = besti[t + 1]; besti[t + 1] = ti;
            }
          }
        }
      }
    }
    float* tmpp = cur; cur = nxt; nxt = tmpp;
  }

  if (tid < 64) {
    int* op = idxout + (size_t)(b * NPTS + rb + tid) * KNN;
#pragma unroll
    for (int k = 0; k < KNN; k++) op[k] = besti[KNN - 1 - k];
  }
}

// ---------------- EdgeConv: gather -> [xi, xj-xi] -> W1(relu) -> W2 -> max_k ----------------
// Block = 64 threads (2 waves), one point per wave. bf16 WMMA, M padded 30->32.
// Weights DMA'd into LDS (async) while the neighbor gather builds E.
template <int CIN, int STRIDE, int KPAD>
__global__ void __launch_bounds__(64) edgeconv_kernel(
    const float* __restrict__ x, const int* __restrict__ nbr,
    const __bf16* __restrict__ w1T, const float* __restrict__ b1,
    const __bf16* __restrict__ w2T, const float* __restrict__ b2,
    float* __restrict__ y) {
  __shared__ alignas(16) __bf16 ldsW1T[64 * KPAD];
  __shared__ alignas(16) __bf16 ldsW2T[64 * 64];
  __shared__ alignas(16) __bf16 ldsE[2][32 * KPAD];
  __shared__ alignas(16) __bf16 ldsH1[2][32 * 64];

  const int tid = threadIdx.x;
  const int wave = tid >> 5, lane = tid & 31;
  const int hi = lane >> 4, l15 = lane & 15;
  const int p = blockIdx.x * 2 + wave;

  // issue weight DMA first; it overlaps the gather below
  {
    const unsigned l1 = lds_addr(ldsW1T);
    for (int t = tid * 8; t < 64 * KPAD; t += 64 * 8) async_copy_b128(l1 + t * 2, w1T + t);
    const unsigned l2 = lds_addr(ldsW2T);
    for (int t = tid * 8; t < 64 * 64; t += 64 * 8) async_copy_b128(l2 + t * 2, w2T + t);
  }

  // build the 32 x KPAD edge matrix (rows 30,31 duplicate edge 0: max-invariant)
  {
    const int e = (lane < KNN) ? lane : 0;
    const int j = nbr[(size_t)p * KNN + e];
    const int bq = p / NPTS;
    const float* xi = x + (size_t)p * STRIDE;
    const float* xj = x + ((size_t)bq * NPTS + j) * STRIDE;
    __bf16* er = &ldsE[wave][lane * KPAD];
#pragma unroll
    for (int c = 0; c < CIN; c++) {
      float vi = xi[c], vj = xj[c];
      er[c] = (__bf16)vi;
      er[CIN + c] = (__bf16)(vj - vi);
    }
#pragma unroll
    for (int c = 2 * CIN; c < KPAD; c++) er[c] = (__bf16)0.0f;
  }
  async_wait0();
  __syncthreads();

  // layer 1: H1 = relu(E @ W1 + b1), 32x64
#pragma unroll
  for (int mt = 0; mt < 2; mt++) {
#pragma unroll
    for (int nt = 0; nt < 4; nt++) {
      v8f acc = {};
#pragma unroll
      for (int ks = 0; ks < KPAD / 32; ks++) {
        ABf16 a, bm;
        const __bf16* ea = &ldsE[wave][(mt * 16 + l15) * KPAD + ks * 32];
        a.h[0] = *(const v8bf*)&ea[8 * hi];
        a.h[1] = *(const v8bf*)&ea[16 + 8 * hi];
        const __bf16* wb = &ldsW1T[(nt * 16 + l15) * KPAD + ks * 32];
        bm.h[0] = *(const v8bf*)&wb[8 * hi];
        bm.h[1] = *(const v8bf*)&wb[16 + 8 * hi];
        acc = wmma_bf16(a.v, bm.v, acc);
      }
#pragma unroll
      for (int r = 0; r < 8; r++) {
        int row = mt * 16 + r + 8 * hi;
        int col = nt * 16 + l15;
        float h = acc[r] + b1[col];
        ldsH1[wave][row * 64 + col] = (__bf16)(h > 0.f ? h : 0.f);
      }
    }
  }
  __syncthreads();

  // layer 2 + column max over the 32 edge rows
#pragma unroll
  for (int nt = 0; nt < 4; nt++) {
    v8f accA = {}, accB = {};
#pragma unroll
    for (int ks = 0; ks < 2; ks++) {
      ABf16 a0, a1, bm;
      const __bf16* wb = &ldsW2T[(nt * 16 + l15) * 64 + ks * 32];
      bm.h[0] = *(const v8bf*)&wb[8 * hi];
      bm.h[1] = *(const v8bf*)&wb[16 + 8 * hi];
      const __bf16* h0 = &ldsH1[wave][l15 * 64 + ks * 32];
      a0.h[0] = *(const v8bf*)&h0[8 * hi];
      a0.h[1] = *(const v8bf*)&h0[16 + 8 * hi];
      const __bf16* h1p = &ldsH1[wave][(16 + l15) * 64 + ks * 32];
      a1.h[0] = *(const v8bf*)&h1p[8 * hi];
      a1.h[1] = *(const v8bf*)&h1p[16 + 8 * hi];
      accA = wmma_bf16(a0.v, bm.v, accA);
      accB = wmma_bf16(a1.v, bm.v, accB);
    }
    float pm = -3.0e38f;
#pragma unroll
    for (int r = 0; r < 8; r++) pm = fmaxf(pm, fmaxf(accA[r], accB[r]));
    pm = fmaxf(pm, __shfl_xor(pm, 16, 32));  // merge lane halves (rows r+8 groups)
    if (lane < 16) {
      int col = nt * 16 + lane;
      y[(size_t)p * 64 + col] = pm + b2[col];
    }
  }
}

// ---------------- fused MLP head: 192->1024->256->128->13 + log_softmax ----------------
// Block = 256 threads (8 waves), 16 points per block. All intermediates in LDS (bf16).
__global__ void __launch_bounds__(256) mlp_head_kernel(
    const float* __restrict__ x1, const float* __restrict__ x2, const float* __restrict__ x3,
    const __bf16* __restrict__ w1T, const float* __restrict__ b1,
    const __bf16* __restrict__ w2T, const float* __restrict__ b2,
    const __bf16* __restrict__ w3T, const float* __restrict__ b3,
    const float* __restrict__ w4, const float* __restrict__ b4,
    float* __restrict__ out) {
  __shared__ alignas(16) __bf16 ldsX[16 * 192];
  __shared__ alignas(16) __bf16 ldsH1[16 * 1024];
  __shared__ alignas(16) __bf16 ldsH2[16 * 256];
  __shared__ alignas(16) __bf16 ldsH3[16 * 128];

  const int tid = threadIdx.x;
  const int wave = tid >> 5, lane = tid & 31;
  const int hi = lane >> 4, l15 = lane & 15;
  const int g0 = blockIdx.x * 16;

  for (int t = tid; t < 16 * 192; t += 256) {
    int mm = t / 192, c = t % 192;
    size_t g = (size_t)(g0 + mm);
    float v = (c < 64) ? x1[g * 64 + c] : (c < 128) ? x2[g * 64 + (c - 64)] : x3[g * 64 + (c - 128)];
    ldsX[t] = (__bf16)v;
  }
  __syncthreads();

  // layer 1: 192 -> 1024, relu (64 N-tiles, 8 per wave); weights streamed from L2
  for (int nt = wave; nt < 64; nt += 8) {
    v8f acc = {};
#pragma unroll
    for (int ks = 0; ks < 6; ks++) {
      ABf16 a, bm;
      const __bf16* xa = &ldsX[l15 * 192 + ks * 32];
      a.h[0] = *(const v8bf*)&xa[8 * hi];
      a.h[1] = *(const v8bf*)&xa[16 + 8 * hi];
      const __bf16* wb = &w1T[(size_t)(nt * 16 + l15) * 192 + ks * 32];
      bm.h[0] = *(const v8bf*)&wb[8 * hi];
      bm.h[1] = *(const v8bf*)&wb[16 + 8 * hi];
      acc = wmma_bf16(a.v, bm.v, acc);
    }
#pragma unroll
    for (int r = 0; r < 8; r++) {
      int row = r + 8 * hi, col = nt * 16 + l15;
      float h = acc[r] + b1[col];
      ldsH1[row * 1024 + col] = (__bf16)(h > 0.f ? h : 0.f);
    }
  }
  __syncthreads();

  // layer 2: 1024 -> 256, relu (16 N-tiles, 2 per wave, 32 k-steps)
  for (int nt = wave; nt < 16; nt += 8) {
    v8f acc = {};
    for (int ks = 0; ks < 32; ks++) {
      ABf16 a, bm;
      const __bf16* xa = &ldsH1[l15 * 1024 + ks * 32];
      a.h[0] = *(const v8bf*)&xa[8 * hi];
      a.h[1] = *(const v8bf*)&xa[16 + 8 * hi];
      const __bf16* wb = &w2T[(size_t)(nt * 16 + l15) * 1024 + ks * 32];
      bm.h[0] = *(const v8bf*)&wb[8 * hi];
      bm.h[1] = *(const v8bf*)&wb[16 + 8 * hi];
      acc = wmma_bf16(a.v, bm.v, acc);
    }
#pragma unroll
    for (int r = 0; r < 8; r++) {
      int row = r + 8 * hi, col = nt * 16 + l15;
      float h = acc[r] + b2[col];
      ldsH2[row * 256 + col] = (__bf16)(h > 0.f ? h : 0.f);
    }
  }
  __syncthreads();

  // layer 3: 256 -> 128, relu (8 N-tiles, 1 per wave, 8 k-steps)
  {
    int nt = wave;
    v8f acc = {};
#pragma unroll
    for (int ks = 0; ks < 8; ks++) {
      ABf16 a, bm;
      const __bf16* xa = &ldsH2[l15 * 256 + ks * 32];
      a.h[0] = *(const v8bf*)&xa[8 * hi];
      a.h[1] = *(const v8bf*)&xa[16 + 8 * hi];
      const __bf16* wb = &w3T[(size_t)(nt * 16 + l15) * 256 + ks * 32];
      bm.h[0] = *(const v8bf*)&wb[8 * hi];
      bm.h[1] = *(const v8bf*)&wb[16 + 8 * hi];
      acc = wmma_bf16(a.v, bm.v, acc);
    }
#pragma unroll
    for (int r = 0; r < 8; r++) {
      int row = r + 8 * hi, col = nt * 16 + l15;
      float h = acc[r] + b3[col];
      ldsH3[row * 128 + col] = (__bf16)(h > 0.f ? h : 0.f);
    }
  }
  __syncthreads();

  // layer 4 (128 -> 13) + log_softmax, one thread per point
  if (tid < 16) {
    const int mrow = tid;
    const size_t g = (size_t)(g0 + mrow);
    float lg[13];
#pragma unroll
    for (int c = 0; c < 13; c++) {
      float s = b4[c];
      for (int k = 0; k < 128; k++) s += (float)ldsH3[mrow * 128 + k] * w4[k * 13 + c];
      lg[c] = s;
    }
    float mx = lg[0];
#pragma unroll
    for (int c = 1; c < 13; c++) mx = fmaxf(mx, lg[c]);
    float sum = 0.f;
#pragma unroll
    for (int c = 0; c < 13; c++) sum += __expf(lg[c] - mx);
    float lse = mx + __logf(sum);
#pragma unroll
    for (int c = 0; c < 13; c++) out[g * 13 + c] = lg[c] - lse;
  }
}

// ---------------- orchestration ----------------

extern "C" void kernel_launch(void* const* d_in, const int* in_sizes, int n_in,
                              void* d_out, int out_size, void* d_ws, size_t ws_size,
                              hipStream_t stream) {
  (void)in_sizes; (void)n_in; (void)out_size; (void)ws_size;
  const float* pts  = (const float*)d_in[0];
  const float* c1w1 = (const float*)d_in[1];  const float* c1b1 = (const float*)d_in[2];
  const float* c1w2 = (const float*)d_in[3];  const float* c1b2 = (const float*)d_in[4];
  const float* c2w1 = (const float*)d_in[5];  const float* c2b1 = (const float*)d_in[6];
  const float* c2w2 = (const float*)d_in[7];  const float* c2b2 = (const float*)d_in[8];
  const float* c3w1 = (const float*)d_in[9];  const float* c3b1 = (const float*)d_in[10];
  const float* c3w2 = (const float*)d_in[11]; const float* c3b2 = (const float*)d_in[12];
  const float* mw1  = (const float*)d_in[13]; const float* mb1  = (const float*)d_in[14];
  const float* mw2  = (const float*)d_in[15]; const float* mb2  = (const float*)d_in[16];
  const float* mw3  = (const float*)d_in[17]; const float* mb3  = (const float*)d_in[18];
  const float* mw4  = (const float*)d_in[19]; const float* mb4  = (const float*)d_in[20];
  float* out = (float*)d_out;

  char* ws = (char*)d_ws;
  size_t off = 0;
  auto alloc = [&](size_t bytes) -> char* {
    char* p = ws + off;
    off = (off + bytes + 255) & ~(size_t)255;
    return p;
  };
  float* xpad8 = (float*)alloc((size_t)BN * 8 * 4);
  float* sqbuf = (float*)alloc((size_t)BN * 4);
  int* idxbuf  = (int*)alloc((size_t)BN * KNN * 4);
  float* x1    = (float*)alloc((size_t)BN * 64 * 4);
  float* x2    = (float*)alloc((size_t)BN * 64 * 4);
  float* x3    = (float*)alloc((size_t)BN * 64 * 4);
  __bf16* c1w1T = (__bf16*)alloc(64 * 32 * 2);
  __bf16* c1w2T = (__bf16*)alloc(64 * 64 * 2);
  __bf16* c2w1T = (__bf16*)alloc(64 * 128 * 2);
  __bf16* c2w2T = (__bf16*)alloc(64 * 64 * 2);
  __bf16* c3w1T = (__bf16*)alloc(64 * 128 * 2);
  __bf16* c3w2T = (__bf16*)alloc(64 * 64 * 2);
  __bf16* mw1T  = (__bf16*)alloc((size_t)1024 * 192 * 2);
  __bf16* mw2T  = (__bf16*)alloc((size_t)256 * 1024 * 2);
  __bf16* mw3T  = (__bf16*)alloc((size_t)128 * 256 * 2);

  pad8_sq_kernel<<<BN / 256, 256, 0, stream>>>(pts, xpad8, sqbuf);

  auto conv = [&](const float* s, __bf16* dst, int K, int KP, int N) {
    int tot = N * KP;
    convert_transpose_kernel<<<(tot + 255) / 256, 256, 0, stream>>>(s, dst, K, KP, N);
  };
  conv(c1w1, c1w1T, 12, 32, 64);
  conv(c1w2, c1w2T, 64, 64, 64);
  conv(c2w1, c2w1T, 128, 128, 64);
  conv(c2w2, c2w2T, 64, 64, 64);
  conv(c3w1, c3w1T, 128, 128, 64);
  conv(c3w2, c3w2T, 64, 64, 64);
  conv(mw1, mw1T, 192, 192, 1024);
  conv(mw2, mw2T, 1024, 1024, 256);
  conv(mw3, mw3T, 256, 256, 128);

  dim3 kgrid(NPTS / 64, BB);
  knn_kernel<8><<<kgrid, 128, 0, stream>>>(xpad8, sqbuf, idxbuf);
  edgeconv_kernel<6, 8, 32><<<BN / 2, 64, 0, stream>>>(xpad8, idxbuf, c1w1T, c1b1, c1w2T, c1b2, x1);

  sqnorm64_kernel<<<BN / 256, 256, 0, stream>>>(x1, sqbuf);
  knn_kernel<64><<<kgrid, 128, 0, stream>>>(x1, sqbuf, idxbuf);
  edgeconv_kernel<64, 64, 128><<<BN / 2, 64, 0, stream>>>(x1, idxbuf, c2w1T, c2b1, c2w2T, c2b2, x2);

  sqnorm64_kernel<<<BN / 256, 256, 0, stream>>>(x2, sqbuf);
  knn_kernel<64><<<kgrid, 128, 0, stream>>>(x2, sqbuf, idxbuf);
  edgeconv_kernel<64, 64, 128><<<BN / 2, 64, 0, stream>>>(x2, idxbuf, c3w1T, c3b1, c3w2T, c3b2, x3);

  mlp_head_kernel<<<BN / 16, 256, 0, stream>>>(x1, x2, x3, mw1T, mb1, mw2T, mb2, mw3T, mb3,
                                               mw4, mb4, out);
}
